// RoPEAttention_63024350102311
// MI455X (gfx1250) — matrix operand order repaired and checked by reading, not compiled, not run on hardware
//
#include <hip/hip_runtime.h>

typedef __bf16 bf16_t;
typedef __attribute__((ext_vector_type(16))) __bf16 v16bf;
typedef __attribute__((ext_vector_type(8)))  __bf16 v8bf;
typedef __attribute__((ext_vector_type(8)))  float  v8f;

#define B_   2
#define L_   2048
#define D_   1024
#define H_   16
#define DH_  64
#define BL_  (B_ * L_)   // 4096
#define N3_  (3 * D_)    // 3072
#define TQ_  64          // q rows per attention block (4 waves x 16)

// ---------------------------------------------------------------------------
// WMMA helper (CDNA5 wave32, v_wmma_f32_16x16x32_bf16)
// ---------------------------------------------------------------------------
static __device__ __forceinline__ v8f wmma_bf16(v16bf a, v16bf b, v8f c) {
  return __builtin_amdgcn_wmma_f32_16x16x32_bf16(false, a, false, b, (short)0, c,
                                                 false, false);
}

static __device__ __forceinline__ v16bf cat8(v8bf lo, v8bf hi) {
  return __builtin_shufflevector(lo, hi, 0, 1, 2, 3, 4, 5, 6, 7,
                                 8, 9, 10, 11, 12, 13, 14, 15);
}

// A operand: 16(M) x 32(K) bf16, rows contiguous with `stride` elements.
// lanes 0-15: M=lane, K {0..7,16..23}; lanes 16-31: M=lane-16, K {8..15,24..31}.
static __device__ __forceinline__ v16bf load_a32(const bf16_t* p, int stride) {
  int lane = threadIdx.x & 31;
  const bf16_t* q = p + (lane & 15) * stride + ((lane >> 4) << 3);
  return cat8(*(const v8bf*)(q), *(const v8bf*)(q + 16));
}

// B operand: 32(K) x 16(N) bf16 stored as [N][K].
// lanes 0-15: N=lane, K {0..7,16..23}; lanes 16-31: N=lane-16, K {8..15,24..31}.
static __device__ __forceinline__ v16bf load_b32(const bf16_t* p, int stride) {
  int lane = threadIdx.x & 31;
  const bf16_t* q = p + (lane & 15) * stride + ((lane >> 4) << 3);
  return cat8(*(const v8bf*)(q), *(const v8bf*)(q + 16));
}

// ---------------------------------------------------------------------------
// Conversion / layout kernels
// ---------------------------------------------------------------------------
__global__ void k_cvt_bf16(const float* __restrict__ src, bf16_t* __restrict__ dst, int n) {
  int i = blockIdx.x * blockDim.x + threadIdx.x;
  if (i < n) dst[i] = (bf16_t)src[i];
}

__global__ void k_transpose_bf16(const float* __restrict__ w, bf16_t* __restrict__ wt,
                                 int rows, int cols) {
  int i = blockIdx.x * blockDim.x + threadIdx.x;
  if (i < rows * cols) {
    int r = i / cols, c = i % cols;
    wt[c * rows + r] = (bf16_t)w[i];
  }
}

// ---------------------------------------------------------------------------
// Shared-B GEMM mainloop: block = 128 threads (4 waves), tile M=64/N=64.
// B tile (64x32 bf16, 4 KB) staged to LDS via global_load_async_to_lds_b128,
// double-buffered with s_wait_asynccnt. Each wave owns 16 M-rows and a
// register-double-buffered A tile. acc[c] = 16x16 f32 for N sub-tile c.
// ---------------------------------------------------------------------------
static __device__ __forceinline__ void gemm_mainloop(const bf16_t* __restrict__ aRow,
                                                     const bf16_t* __restrict__ bBase,
                                                     bf16_t* shB,  // [2][64*32]
                                                     int wv, int lane, v8f acc[4]) {
  auto issueB = [&](int k0, int buf) {
#pragma unroll
    for (int i = 0; i < 2; ++i) {
      int c = wv * 2 + i;                    // chunk 0..7 (8 rows each)
      int n = c * 8 + (lane >> 2);           // B row 0..63
      unsigned lds = (unsigned)(uintptr_t)(shB + buf * (64 * 32)) + c * 512 + lane * 16;
      const bf16_t* src = bBase + (size_t)n * D_ + k0 + (lane & 3) * 8;
      unsigned long long ga = (unsigned long long)(uintptr_t)src;
      asm volatile("global_load_async_to_lds_b128 %0, %1, off"
                   :: "v"(lds), "v"(ga) : "memory");
    }
  };

  issueB(0, 0);
  v16bf aCur = load_a32(aRow, D_);
  for (int k0 = 0; k0 < D_; k0 += 32) {
    const int buf = (k0 >> 5) & 1;
    v16bf aNext;
    bool more = (k0 + 32 < D_);
    if (more) {
      issueB(k0 + 32, buf ^ 1);              // overlap next B-tile DMA
      aNext = load_a32(aRow + k0 + 32, D_);  // prefetch next A into regs
      asm volatile("s_wait_asynccnt 0x2" ::: "memory");
    } else {
      asm volatile("s_wait_asynccnt 0x0" ::: "memory");
    }
    __syncthreads();
    const bf16_t* bb = shB + buf * (64 * 32);
#pragma unroll
    for (int c = 0; c < 4; ++c) {
      acc[c] = wmma_bf16(aCur, load_b32(bb + c * 16 * 32, 32), acc[c]);
    }
    __syncthreads();
    if (more) aCur = aNext;
  }
}

// ---------------------------------------------------------------------------
// QKV GEMM with fused RoPE + layout scatter.
// grid = (BL/64, 3072/64), block = 128.
// ---------------------------------------------------------------------------
__global__ __launch_bounds__(128) void k_qkv_gemm(const bf16_t* __restrict__ xb,
                                                  const bf16_t* __restrict__ wt,
                                                  bf16_t* __restrict__ Qb,
                                                  bf16_t* __restrict__ Kb,
                                                  bf16_t* __restrict__ Vt) {
  const int lane = threadIdx.x & 31;
  const int wv = threadIdx.x >> 5;
  const int m0 = blockIdx.x * 64 + wv * 16;
  const int n0 = blockIdx.y * 64;
  __shared__ __align__(16) bf16_t shB[2][64 * 32];

  v8f acc[4] = {};
  gemm_mainloop(xb + m0 * D_, wt + n0 * D_, &shB[0][0], wv, lane, acc);

  const int col = lane & 15;
  const int hig = lane >> 4;
  const int s = n0 >> 10;            // 0=q, 1=k, 2=v
  const int h = (n0 & 1023) >> 6;

#pragma unroll
  for (int r = 0; r < 8; ++r) {
    int row = m0 + hig * 8 + r;
    int b = row >> 11;
    int l = row & 2047;
    if (s < 2) {
      bf16_t* dst = (s == 0 ? Qb : Kb) + ((b * H_ + h) * L_ + l) * DH_;
      float lpos = (float)l;
#pragma unroll
      for (int c = 0; c < 2; ++c) {
        int d = c * 16 + col;  // 0..31
        float fr = lpos * __expf((float)d * -0.28782313662425574f);
        float cs = __cosf(fr), sn = __sinf(fr);
        float xv = acc[c][r];
        float yv = acc[c + 2][r];
        dst[d]      = (bf16_t)(xv * cs - yv * sn);
        dst[d + 32] = (bf16_t)(yv * cs + xv * sn);
      }
    } else {
      bf16_t* dst = Vt + ((b * H_ + h) * DH_) * L_ + l;
#pragma unroll
      for (int c = 0; c < 4; ++c) {
        int d = c * 16 + col;
        dst[d * L_] = (bf16_t)acc[c][r];
      }
    }
  }
}

// ---------------------------------------------------------------------------
// Flash attention: 4 waves/block share K/V tiles staged into LDS with
// global_load_async_to_lds_b128 (ASYNCcnt), double-buffered.
// grid = (L/64, B*H), block = 128. Each wave owns 16 q-rows.
// ---------------------------------------------------------------------------
__global__ __launch_bounds__(128) void k_attn(const bf16_t* __restrict__ Qb,
                                              const bf16_t* __restrict__ Kb,
                                              const bf16_t* __restrict__ Vt,
                                              const int* __restrict__ key_mask,
                                              bf16_t* __restrict__ Ob) {
  const int tid = threadIdx.x;
  const int lane = tid & 31;
  const int wv = tid >> 5;
  const int col = lane & 15;
  const int hig = lane >> 4;
  const int bh = blockIdx.y;
  const int b = bh >> 4;
  const int h = bh & 15;
  const int m0 = blockIdx.x * TQ_ + wv * 16;

  __shared__ __align__(16) bf16_t shK[2][32 * DH_];  // [key][dh]
  __shared__ __align__(16) bf16_t shV[2][DH_ * 32];  // [dh][key]
  __shared__ __align__(16) bf16_t shP[4][16 * 32];   // per-wave P swizzle

  const bf16_t* Qp = Qb + (bh * L_ + m0) * DH_;
  const char* Kg = (const char*)(Kb + (size_t)bh * L_ * DH_);
  const char* Vg = (const char*)(Vt + (size_t)bh * DH_ * L_);
  const int* mk = key_mask + b * L_;

  v16bf qa0 = load_a32(Qp, DH_);
  v16bf qa1 = load_a32(Qp + 32, DH_);

  v8f o0 = {}, o1 = {}, o2 = {}, o3 = {};
  float mrow[8], lrow[8];
#pragma unroll
  for (int r = 0; r < 8; ++r) { mrow[r] = -3.0e38f; lrow[r] = 0.0f; }
  const float scale = 0.125f;

  auto issue = [&](int n0, int buf) {
#pragma unroll
    for (int i = 0; i < 4; ++i) {
      int c = wv * 4 + i;            // 0..15
      unsigned lds;
      const char* src;
      if (c < 8) {                   // K tile: 32 keys x 128 B, contiguous
        lds = (unsigned)(uintptr_t)(&shK[buf][0]) + c * 512 + lane * 16;
        src = Kg + n0 * (DH_ * 2) + c * 512 + lane * 16;
      } else {                       // V tile: 64 dh rows x 64 B (strided rows)
        int cv = c - 8;
        int dh = cv * 8 + (lane >> 2);
        lds = (unsigned)(uintptr_t)(&shV[buf][0]) + cv * 512 + lane * 16;
        src = Vg + (size_t)dh * (L_ * 2) + n0 * 2 + (lane & 3) * 16;
      }
      unsigned long long ga = (unsigned long long)(uintptr_t)src;
      asm volatile("global_load_async_to_lds_b128 %0, %1, off"
                   :: "v"(lds), "v"(ga) : "memory");
    }
  };

  issue(0, 0);
  for (int n0 = 0; n0 < L_; n0 += 32) {
    const int buf = (n0 >> 5) & 1;
    if (n0 + 32 < L_) {
      issue(n0 + 32, buf ^ 1);
      asm volatile("s_wait_asynccnt 0x4" ::: "memory");
    } else {
      asm volatile("s_wait_asynccnt 0x0" ::: "memory");
    }
    __syncthreads();

    const bf16_t* kbuf = &shK[buf][0];
    const bf16_t* vbuf = &shV[buf][0];

    v8f s0 = {}, s1 = {};
    s0 = wmma_bf16(qa0, load_b32(kbuf, DH_), s0);
    s0 = wmma_bf16(qa1, load_b32(kbuf + 32, DH_), s0);
    s1 = wmma_bf16(qa0, load_b32(kbuf + 16 * DH_, DH_), s1);
    s1 = wmma_bf16(qa1, load_b32(kbuf + 16 * DH_ + 32, DH_), s1);

    float a0 = mk[n0 + col]      ? 0.0f : -3.0e38f;
    float a1 = mk[n0 + 16 + col] ? 0.0f : -3.0e38f;

    float p0[8], p1[8], corr[8];
#pragma unroll
    for (int r = 0; r < 8; ++r) {
      float v0 = s0[r] * scale + a0;
      float v1 = s1[r] * scale + a1;
      float mx = fmaxf(v0, v1);
      mx = fmaxf(mx, __shfl_xor(mx, 1));
      mx = fmaxf(mx, __shfl_xor(mx, 2));
      mx = fmaxf(mx, __shfl_xor(mx, 4));
      mx = fmaxf(mx, __shfl_xor(mx, 8));
      float nm = fmaxf(mrow[r], mx);
      corr[r] = __expf(mrow[r] - nm);
      float e0 = __expf(v0 - nm);
      float e1 = __expf(v1 - nm);
      p0[r] = e0; p1[r] = e1;
      float sum = e0 + e1;
      sum += __shfl_xor(sum, 1);
      sum += __shfl_xor(sum, 2);
      sum += __shfl_xor(sum, 4);
      sum += __shfl_xor(sum, 8);
      lrow[r] = lrow[r] * corr[r] + sum;
      mrow[r] = nm;
    }
#pragma unroll
    for (int r = 0; r < 8; ++r) {
      o0[r] *= corr[r]; o1[r] *= corr[r]; o2[r] *= corr[r]; o3[r] *= corr[r];
    }

#pragma unroll
    for (int r = 0; r < 8; ++r) {
      int row = hig * 8 + r;
      shP[wv][row * 32 + col]      = (bf16_t)p0[r];
      shP[wv][row * 32 + 16 + col] = (bf16_t)p1[r];
    }
    v16bf pa = load_a32(&shP[wv][0], 32);

    o0 = wmma_bf16(pa, load_b32(vbuf + 0 * 16 * 32, 32), o0);
    o1 = wmma_bf16(pa, load_b32(vbuf + 1 * 16 * 32, 32), o1);
    o2 = wmma_bf16(pa, load_b32(vbuf + 2 * 16 * 32, 32), o2);
    o3 = wmma_bf16(pa, load_b32(vbuf + 3 * 16 * 32, 32), o3);
    __syncthreads();
  }

#pragma unroll
  for (int r = 0; r < 8; ++r) {
    int l = m0 + hig * 8 + r;
    float inv = 1.0f / lrow[r];
    bf16_t* dst = Ob + (b * L_ + l) * D_ + h * DH_;
    dst[0 * 16 + col] = (bf16_t)(o0[r] * inv);
    dst[1 * 16 + col] = (bf16_t)(o1[r] * inv);
    dst[2 * 16 + col] = (bf16_t)(o2[r] * inv);
    dst[3 * 16 + col] = (bf16_t)(o3[r] * inv);
  }
}

// ---------------------------------------------------------------------------
// Output projection GEMM -> f32 d_out. grid = (BL/64, D/64), block = 128.
// ---------------------------------------------------------------------------
__global__ __launch_bounds__(128) void k_proj_gemm(const bf16_t* __restrict__ Ob,
                                                   const bf16_t* __restrict__ wt,
                                                   float* __restrict__ out) {
  const int lane = threadIdx.x & 31;
  const int wv = threadIdx.x >> 5;
  const int m0 = blockIdx.x * 64 + wv * 16;
  const int n0 = blockIdx.y * 64;
  __shared__ __align__(16) bf16_t shB[2][64 * 32];

  v8f acc[4] = {};
  gemm_mainloop(Ob + m0 * D_, wt + n0 * D_, &shB[0][0], wv, lane, acc);

  const int col = lane & 15;
  const int hig = lane >> 4;
#pragma unroll
  for (int r = 0; r < 8; ++r) {
    int row = m0 + hig * 8 + r;
    float* dst = out + row * D_ + n0;
#pragma unroll
    for (int c = 0; c < 4; ++c) dst[c * 16 + col] = acc[c][r];
  }
}

// ---------------------------------------------------------------------------
extern "C" void kernel_launch(void* const* d_in, const int* in_sizes, int n_in,
                              void* d_out, int out_size, void* d_ws, size_t ws_size,
                              hipStream_t stream) {
  (void)in_sizes; (void)n_in; (void)out_size; (void)ws_size;
  const float* x        = (const float*)d_in[0];
  const int*   key_mask = (const int*)d_in[1];
  const float* w_qkv    = (const float*)d_in[2];
  const float* w_proj   = (const float*)d_in[3];
  float* out = (float*)d_out;

  char* ws = (char*)d_ws;
  size_t off = 0;
  auto alloc = [&](size_t bytes) -> void* {
    void* p = ws + off;
    off += (bytes + 255) & ~(size_t)255;
    return p;
  };
  bf16_t* xb     = (bf16_t*)alloc((size_t)BL_ * D_ * 2);
  bf16_t* wqkvT  = (bf16_t*)alloc((size_t)N3_ * D_ * 2);
  bf16_t* wprojT = (bf16_t*)alloc((size_t)D_ * D_ * 2);
  bf16_t* Qb     = (bf16_t*)alloc((size_t)B_ * H_ * L_ * DH_ * 2);
  bf16_t* Kb     = (bf16_t*)alloc((size_t)B_ * H_ * L_ * DH_ * 2);
  bf16_t* Vt     = (bf16_t*)alloc((size_t)B_ * H_ * L_ * DH_ * 2);
  bf16_t* Ob     = (bf16_t*)alloc((size_t)BL_ * D_ * 2);

  k_cvt_bf16<<<(BL_ * D_ + 255) / 256, 256, 0, stream>>>(x, xb, BL_ * D_);
  k_transpose_bf16<<<(D_ * N3_ + 255) / 256, 256, 0, stream>>>(w_qkv, wqkvT, D_, N3_);
  k_transpose_bf16<<<(D_ * D_ + 255) / 256, 256, 0, stream>>>(w_proj, wprojT, D_, D_);

  k_qkv_gemm<<<dim3(BL_ / 64, N3_ / 64), 128, 0, stream>>>(xb, wqkvT, Qb, Kb, Vt);
  k_attn<<<dim3(L_ / TQ_, B_ * H_), 128, 0, stream>>>(Qb, Kb, Vt, key_mask, Ob);
  k_proj_gemm<<<dim3(BL_ / 64, D_ / 64), 128, 0, stream>>>(Ob, wprojT, out);
}
